// MyDecoderWithAttention_47407849013552
// MI455X (gfx1250) — matrix operand order repaired and checked by reading, not compiled, not run on hardware
//
#include <hip/hip_runtime.h>
#include <hip/hip_bf16.h>
#include <stdint.h>

// ---------------------------------------------------------------------------
// Show-Attend-Tell decoder for MI455X (gfx1250), bf16 WMMA everywhere.
// B=128, T=96, E=D=A=EMB=512, V=4096.
// ---------------------------------------------------------------------------

typedef __bf16 bf16_t;
typedef __attribute__((ext_vector_type(16))) __bf16 v16bf;
typedef __attribute__((ext_vector_type(8)))  float  v8f;

#define NB   128   // batch
#define NT   96    // time steps
#define NE   512   // encoder dim = attention dim = decoder dim = emb dim
#define NV   4096  // vocab
#define NG   2048  // 4*D (LSTM gates)
#define NX   1024  // EMB + E (LSTM input)

__device__ __forceinline__ float sigmf_(float x) { return 1.0f / (1.0f + expf(-x)); }

__device__ __forceinline__ v8f wmma_bf16(v16bf a, v16bf b, v8f c) {
  // D = A(16x32 bf16) x B(32x16 bf16) + C(16x16 f32)
  return __builtin_amdgcn_wmma_f32_16x16x32_bf16(false, a, false, b, (short)0, c,
                                                 false, false);
}

// A-operand (16x32 tile) from row-major bf16 buffer sh[16][ldk], tile k-origin k0.
// Layout: lanes 0-15: M=lane, K=0..15; lanes 16-31: M=lane-16, K=16..31 (matches load_B).
__device__ __forceinline__ v16bf load_A(const bf16_t* sh, int ldk, int k0, int lane) {
  int half = lane >> 4;
  const bf16_t* row = sh + (lane & 15) * ldk + k0;
  int b1 = half ? 16 : 0;
  int b2 = half ? 24 : 8;
  union { v16bf v; uint32_t u[8]; } r;
  const uint32_t* p1 = (const uint32_t*)(row + b1);
  const uint32_t* p2 = (const uint32_t*)(row + b2);
#pragma unroll
  for (int i = 0; i < 4; ++i) { r.u[i] = p1[i]; r.u[4 + i] = p2[i]; }
  return r.v;
}

// B-operand from WMMA-packed global layout Bp[kt][N][32]:
// lane n (0-15) needs B[k0..k0+15][n], lane n+16 needs B[k0+16..k0+31][n] -> 32B contiguous.
__device__ __forceinline__ v16bf load_B(const bf16_t* Bp, int N, int kt, int n0, int lane) {
  const bf16_t* p = Bp + (((size_t)kt * N + (size_t)(n0 + (lane & 15))) << 5)
                       + ((lane >> 4) << 4);
  union { v16bf v; uint4 q[2]; } r;
  const uint4* s = (const uint4*)p;
  r.q[0] = s[0]; r.q[1] = s[1];
  return r.v;
}

// Address of a lane's B-operand bytes (for global_prefetch warming).
__device__ __forceinline__ const bf16_t* addr_B(const bf16_t* Bp, int N, int kt, int n0,
                                                int lane) {
  return Bp + (((size_t)kt * N + (size_t)(n0 + (lane & 15))) << 5) + ((lane >> 4) << 4);
}

// ---------------------------------------------------------------------------
// Setup kernels
// ---------------------------------------------------------------------------

// Stable descending argsort of caption lengths (matches jnp.argsort(-lens)).
__global__ void k_order(const int* __restrict__ lens, int* __restrict__ order,
                        int* __restrict__ lens_s) {
  int i = threadIdx.x;            // 0..127
  int li = lens[i];
  int rank = 0;
  for (int j = 0; j < NB; ++j) {
    int lj = lens[j];
    if (lj > li || (lj == li && j < i)) ++rank;
  }
  order[rank]  = i;
  lens_s[rank] = li;
}

// Gather sorted encoder rows, produce f32 + bf16 copies.
__global__ void k_gather(const float* __restrict__ enc, const int* __restrict__ order,
                         float* __restrict__ enc_s, bf16_t* __restrict__ enc_sbf) {
  int r = blockIdx.x;
  int src = order[r];
  for (int e = threadIdx.x; e < NE; e += blockDim.x) {
    float v = enc[(size_t)src * NE + e];
    enc_s[(size_t)r * NE + e]   = v;
    enc_sbf[(size_t)r * NE + e] = (bf16_t)v;
  }
}

// Pack W[N][K] (row-major, used as B[k][n] = W[n][k]) -> Bp[kt][n][32] bf16.
__global__ void k_pack_nk(const float* __restrict__ W, bf16_t* __restrict__ Bp,
                          int N, int K) {
  size_t idx = (size_t)blockIdx.x * 256 + threadIdx.x;
  size_t total = (size_t)N * K;
  if (idx >= total) return;
  int jj = (int)(idx & 31);
  size_t r = idx >> 5;
  int n  = (int)(r % N);
  int kt = (int)(r / N);
  Bp[idx] = (bf16_t)W[(size_t)n * K + (size_t)kt * 32 + jj];
}

// Pack B[K][N] (row-major) -> Bp[kt][n][32] bf16 (used for enc in awe GEMM).
__global__ void k_pack_kn(const float* __restrict__ Bsrc, bf16_t* __restrict__ Bp,
                          int N, int K) {
  size_t idx = (size_t)blockIdx.x * 256 + threadIdx.x;
  size_t total = (size_t)N * K;
  if (idx >= total) return;
  int jj = (int)(idx & 31);
  size_t r = idx >> 5;
  int n  = (int)(r % N);
  int kt = (int)(r / N);
  Bp[idx] = (bf16_t)Bsrc[(size_t)(kt * 32 + jj) * N + n];
}

// C[128][N] = A(128x512 bf16) @ Bp + bias ; optionally f32 and/or bf16 outputs.
// grid (8, N/128), 256 threads (8 waves, 1 tile each).
__global__ void __launch_bounds__(256) k_gemm_init(
    const bf16_t* __restrict__ Abf, const bf16_t* __restrict__ Bp,
    const float* __restrict__ bias, float* __restrict__ outF,
    bf16_t* __restrict__ outB, int N) {
  __shared__ bf16_t sh_a[16 * NE];
  int tid = threadIdx.x;
  int m0 = blockIdx.x * 16;
  const uint32_t* src = (const uint32_t*)(Abf + (size_t)m0 * NE);
  uint32_t* dst = (uint32_t*)sh_a;
  for (int i = tid; i < 16 * NE / 2; i += 256) dst[i] = src[i];
  __syncthreads();
  int w = tid >> 5, lane = tid & 31;
  int n0 = blockIdx.y * 128 + w * 16;
  v8f acc = {};
#pragma unroll
  for (int kt = 0; kt < NE / 32; ++kt)
    acc = wmma_bf16(load_A(sh_a, NE, kt * 32, lane), load_B(Bp, N, kt, n0, lane), acc);
  int n = n0 + (lane & 15);
  int mb = (lane >> 4) << 3;
  float bv = bias[n];
#pragma unroll
  for (int v = 0; v < 8; ++v) {
    int row = m0 + mb + v;
    float val = acc[v] + bv;
    if (outF) outF[(size_t)row * N + n] = val;
    if (outB) outB[(size_t)row * N + n] = (bf16_t)val;
  }
}

// ---------------------------------------------------------------------------
// Per-timestep fused kernel: attention + gate + LSTM cell.
// grid 8 blocks (16 rows each), 256 threads (8 waves).
// ---------------------------------------------------------------------------
#define K1_SMEM 97920

__global__ void __launch_bounds__(256) k_step(
    const int* __restrict__ order, const int* __restrict__ lens_s,
    const int* __restrict__ caps, const float* __restrict__ emb,
    const float* __restrict__ att1,
    bf16_t* __restrict__ hbf, float* __restrict__ cbuf,
    const bf16_t* __restrict__ Wdec_p, const float* __restrict__ b_dec,
    const float* __restrict__ Wfull, const float* __restrict__ b_full,
    const bf16_t* __restrict__ enc_p,
    const bf16_t* __restrict__ Wfb_p, const float* __restrict__ b_fbeta,
    const bf16_t* __restrict__ Wih_p, const float* __restrict__ b_ih,
    const bf16_t* __restrict__ Whh_p, const float* __restrict__ b_hh,
    int t) {
  extern __shared__ char smem[];
  bf16_t* sh_h      = (bf16_t*)(smem);              // 16x512 bf16  (16384)
  bf16_t* sh_x      = (bf16_t*)(smem + 16384);      // 16x1024 bf16 (32768) [emb | gated awe]
  float*  sh_att2   = (float*)(smem + 49152);       // 16x512 f32   (32768) (reused as awe)
  float*  sh_scores = (float*)(smem + 81920);       // 16x128 f32   (8192)
  bf16_t* sh_alpha  = (bf16_t*)(smem + 90112);      // 16x128 bf16  (4096)
  float*  sh_wfull  = (float*)(smem + 94208);       // 512 f32      (2048)
  float*  sh_red    = (float*)(smem + 96256);       // 16x16 f32    (1024)
  float*  sh_stat   = (float*)(smem + 97280);       // 16 f32       (64)
  int*    sh_lens   = (int*)(smem + 97344);         // 128 i32      (512)
  int*    sh_cap    = (int*)(smem + 97856);         // 16 i32       (64)
  float*  sh_awe    = sh_att2;                      // union (att2 dead after scores)

  int tid = threadIdx.x;
  int m0 = blockIdx.x * 16;
  int w = tid >> 5, lane = tid & 31;

  // ---- prefetch: warm L2/L0 for the first K-slices of this wave's B streams
  //      (global_prefetch_b8; weights are L2-resident, this hides cold-start
  //       latency of each WMMA phase on the step critical path) ----
#pragma unroll
  for (int i = 0; i < 4; ++i) {
    __builtin_prefetch(addr_B(Wdec_p, NE, 0, (w * 4 + i) * 16, lane), 0, 3);
    __builtin_prefetch(addr_B(Wdec_p, NE, 1, (w * 4 + i) * 16, lane), 0, 3);
    __builtin_prefetch(addr_B(Wfb_p,  NE, 0, (w * 4 + i) * 16, lane), 0, 3);
  }
#pragma unroll
  for (int g = 0; g < 4; ++g) {
    __builtin_prefetch(addr_B(Wih_p, NG, 0, g * NE + w * 64, lane), 0, 3);
    __builtin_prefetch(addr_B(Whh_p, NG, 0, g * NE + w * 64, lane), 0, 3);
  }

  // ---- phase 0a: stage lens, W_full, h tile, caps ----
  if (tid < NB) sh_lens[tid] = lens_s[tid];
  if (tid < 16) sh_cap[tid] = caps[(size_t)order[m0 + tid] * NT + t];
  for (int i = tid; i < NE; i += 256) sh_wfull[i] = Wfull[i];
  {
    const uint32_t* hsrc = (const uint32_t*)(hbf + (size_t)m0 * NE);
    uint32_t* hdst = (uint32_t*)sh_h;
    for (int i = tid; i < 16 * NE / 2; i += 256) hdst[i] = hsrc[i];
  }
  __syncthreads();

  // ---- phase 0b: embedding half of x ----
  for (int i = tid; i < 16 * NE; i += 256) {
    int m = i >> 9, e = i & 511;
    sh_x[m * NX + e] = (bf16_t)emb[(size_t)sh_cap[m] * NE + e];
  }

  // ---- phase 1: att2 = h @ W_dec^T + b_dec (WMMA, 4 tiles/wave) ----
  {
    v8f acc[4] = {};
#pragma unroll
    for (int kt = 0; kt < NE / 32; ++kt) {
      v16bf a = load_A(sh_h, NE, kt * 32, lane);
#pragma unroll
      for (int i = 0; i < 4; ++i)
        acc[i] = wmma_bf16(a, load_B(Wdec_p, NE, kt, (w * 4 + i) * 16, lane), acc[i]);
    }
    int mb = (lane >> 4) << 3;
#pragma unroll
    for (int i = 0; i < 4; ++i) {
      int n = (w * 4 + i) * 16 + (lane & 15);
      float bd = b_dec[n];
#pragma unroll
      for (int v = 0; v < 8; ++v) sh_att2[(mb + v) * NE + n] = acc[i][v] + bd;
    }
  }
  __syncthreads();

  // ---- phase 2: scores + masking + softmax -> alpha (bf16) ----
  {
    float bfull = b_full[0];
    int ti = tid >> 4, tj = tid & 15;
    const float* at2 = sh_att2 + ti * NE;
    for (int jj = 0; jj < 8; ++jj) {
      int j = (jj << 4) + tj;
      const float4* a1 = (const float4*)(att1 + (size_t)j * NE);
      const float4* wf = (const float4*)sh_wfull;
      float s = bfull;
      for (int a4 = 0; a4 < NE / 4; ++a4) {
        float4 u = a1[a4];
        float4 wv = wf[a4];
        const float* c = at2 + a4 * 4;
        s += fmaxf(u.x + c[0], 0.f) * wv.x + fmaxf(u.y + c[1], 0.f) * wv.y +
             fmaxf(u.z + c[2], 0.f) * wv.z + fmaxf(u.w + c[3], 0.f) * wv.w;
      }
      if (t >= sh_lens[j]) s = -1000000000.0f;
      sh_scores[ti * NB + j] = s;
    }
    __syncthreads();
    float pm = -3.0e38f;
    for (int jj = 0; jj < 8; ++jj) pm = fmaxf(pm, sh_scores[ti * NB + (jj << 4) + tj]);
    sh_red[tid] = pm;
    __syncthreads();
    if (tj == 0) {
      float mx = sh_red[ti * 16];
      for (int k = 1; k < 16; ++k) mx = fmaxf(mx, sh_red[ti * 16 + k]);
      sh_stat[ti] = mx;
    }
    __syncthreads();
    float rm = sh_stat[ti];
    float ps = 0.f;
    for (int jj = 0; jj < 8; ++jj) ps += expf(sh_scores[ti * NB + (jj << 4) + tj] - rm);
    sh_red[tid] = ps;
    __syncthreads();
    if (tj == 0) {
      float sm = 0.f;
      for (int k = 0; k < 16; ++k) sm += sh_red[ti * 16 + k];
      sh_stat[ti] = sm;
    }
    __syncthreads();
    float inv = 1.0f / sh_stat[ti];
    for (int jj = 0; jj < 8; ++jj) {
      int j = (jj << 4) + tj;
      sh_alpha[ti * NB + j] = (bf16_t)(expf(sh_scores[ti * NB + j] - rm) * inv);
    }
  }
  __syncthreads();

  // ---- phase 3: awe = alpha @ enc (K=128, WMMA) -> overwrite att2 region ----
  {
    v8f acc[4] = {};
#pragma unroll
    for (int kt = 0; kt < 4; ++kt) {
      v16bf a = load_A(sh_alpha, NB, kt * 32, lane);
#pragma unroll
      for (int i = 0; i < 4; ++i)
        acc[i] = wmma_bf16(a, load_B(enc_p, NE, kt, (w * 4 + i) * 16, lane), acc[i]);
    }
    int mb = (lane >> 4) << 3;
#pragma unroll
    for (int i = 0; i < 4; ++i) {
      int n = (w * 4 + i) * 16 + (lane & 15);
#pragma unroll
      for (int v = 0; v < 8; ++v) sh_awe[(mb + v) * NE + n] = acc[i][v];
    }
  }
  __syncthreads();

  // ---- phase 4: gate = sigmoid(h @ W_fbeta^T + b); x[:,512:] = gate * awe ----
  {
    v8f acc[4] = {};
#pragma unroll
    for (int kt = 0; kt < NE / 32; ++kt) {
      v16bf a = load_A(sh_h, NE, kt * 32, lane);
#pragma unroll
      for (int i = 0; i < 4; ++i)
        acc[i] = wmma_bf16(a, load_B(Wfb_p, NE, kt, (w * 4 + i) * 16, lane), acc[i]);
    }
    int mb = (lane >> 4) << 3;
#pragma unroll
    for (int i = 0; i < 4; ++i) {
      int n = (w * 4 + i) * 16 + (lane & 15);
      float bfb = b_fbeta[n];
#pragma unroll
      for (int v = 0; v < 8; ++v) {
        int m = mb + v;
        float g = sigmf_(acc[i][v] + bfb);
        sh_x[m * NX + NE + n] = (bf16_t)(g * sh_awe[m * NE + n]);
      }
    }
  }
  __syncthreads();

  // ---- phase 5: gates = x@W_ih^T + h@W_hh^T + biases, LSTM cell update ----
  // Wave w owns hidden dims [w*64, w*64+64); per 16-dim slice s it computes the
  // i/f/g/o columns for that slice entirely in accumulator registers.
  for (int s = 0; s < 4; ++s) {
    v8f acc[4] = {};
    for (int kt = 0; kt < 48; ++kt) {
      v16bf a = (kt < 32) ? load_A(sh_x, NX, kt * 32, lane)
                          : load_A(sh_h, NE, (kt - 32) * 32, lane);
      const bf16_t* Bp = (kt < 32) ? Wih_p : Whh_p;
      int ktl = (kt < 32) ? kt : kt - 32;
#pragma unroll
      for (int g = 0; g < 4; ++g)
        acc[g] = wmma_bf16(a, load_B(Bp, NG, ktl, g * NE + w * 64 + s * 16, lane), acc[g]);
    }
    int dl = w * 64 + s * 16 + (lane & 15);
    int mb = (lane >> 4) << 3;
    float bg[4];
#pragma unroll
    for (int g = 0; g < 4; ++g) {
      int n = g * NE + dl;
      bg[g] = b_ih[n] + b_hh[n];
    }
#pragma unroll
    for (int v = 0; v < 8; ++v) {
      int row = m0 + mb + v;
      if (t < sh_lens[row]) {
        size_t idx = (size_t)row * NE + dl;
        float gi = sigmf_(acc[0][v] + bg[0]);
        float gf = sigmf_(acc[1][v] + bg[1]);
        float gg = tanhf(acc[2][v] + bg[2]);
        float go = sigmf_(acc[3][v] + bg[3]);
        float cn = gf * cbuf[idx] + gi * gg;
        float hn = go * tanhf(cn);
        cbuf[idx] = cn;
        hbf[idx] = (bf16_t)hn;
      }
    }
  }
}

// ---------------------------------------------------------------------------
// Vocab projection: preds = h_new @ W_fc^T + b_fc (masked; zeros if invalid).
// grid (8,16), 256 threads; each wave 2 N-tiles, K=512.
// ---------------------------------------------------------------------------
__global__ void __launch_bounds__(256) k_preds(
    const bf16_t* __restrict__ hbf, const bf16_t* __restrict__ Wfc_p,
    const float* __restrict__ b_fc, const int* __restrict__ lens_s,
    float* __restrict__ out, int t) {
  __shared__ bf16_t sh_h[16 * NE];
  __shared__ int sh_lens[16];
  int tid = threadIdx.x;
  int m0 = blockIdx.x * 16;
  int w = tid >> 5, lane = tid & 31;
  int nbase = (blockIdx.y * 16 + w * 2) * 16;
  // warm the first two K-slices of this wave's W_fc stream while staging h
  __builtin_prefetch(addr_B(Wfc_p, NV, 0, nbase, lane), 0, 3);
  __builtin_prefetch(addr_B(Wfc_p, NV, 0, nbase + 16, lane), 0, 3);
  __builtin_prefetch(addr_B(Wfc_p, NV, 1, nbase, lane), 0, 3);
  __builtin_prefetch(addr_B(Wfc_p, NV, 1, nbase + 16, lane), 0, 3);
  {
    const uint32_t* src = (const uint32_t*)(hbf + (size_t)m0 * NE);
    uint32_t* dst = (uint32_t*)sh_h;
    for (int i = tid; i < 16 * NE / 2; i += 256) dst[i] = src[i];
  }
  if (tid < 16) sh_lens[tid] = lens_s[m0 + tid];
  __syncthreads();
  v8f acc[2] = {};
#pragma unroll
  for (int kt = 0; kt < NE / 32; ++kt) {
    v16bf a = load_A(sh_h, NE, kt * 32, lane);
#pragma unroll
    for (int i = 0; i < 2; ++i)
      acc[i] = wmma_bf16(a, load_B(Wfc_p, NV, kt, nbase + i * 16, lane), acc[i]);
  }
  int mb = (lane >> 4) << 3;
#pragma unroll
  for (int i = 0; i < 2; ++i) {
    int n = nbase + i * 16 + (lane & 15);
    float bias = b_fc[n];
#pragma unroll
    for (int v = 0; v < 8; ++v) {
      int m = mb + v;
      int row = m0 + m;
      bool valid = (t < sh_lens[m]);
      out[((size_t)row * NT + t) * NV + n] = valid ? (acc[i][v] + bias) : 0.0f;
    }
  }
}

// ---------------------------------------------------------------------------
// Host driver
// ---------------------------------------------------------------------------
extern "C" void kernel_launch(void* const* d_in, const int* in_sizes, int n_in,
                              void* d_out, int out_size, void* d_ws, size_t ws_size,
                              hipStream_t stream) {
  (void)in_sizes; (void)n_in; (void)out_size; (void)ws_size;
  const float* encoder_out      = (const float*)d_in[0];
  const int*   encoded_captions = (const int*)d_in[1];
  const int*   caption_lengths  = (const int*)d_in[2];
  const float* W_enc    = (const float*)d_in[3];
  const float* b_enc    = (const float*)d_in[4];
  const float* W_dec    = (const float*)d_in[5];
  const float* b_dec    = (const float*)d_in[6];
  const float* W_full   = (const float*)d_in[7];
  const float* b_full   = (const float*)d_in[8];
  const float* W_init_h = (const float*)d_in[9];
  const float* b_init_h = (const float*)d_in[10];
  const float* W_init_c = (const float*)d_in[11];
  const float* b_init_c = (const float*)d_in[12];
  const float* W_fbeta  = (const float*)d_in[13];
  const float* b_fbeta  = (const float*)d_in[14];
  const float* W_fc     = (const float*)d_in[15];
  const float* b_fc     = (const float*)d_in[16];
  const float* emb_table = (const float*)d_in[17];
  const float* W_ih     = (const float*)d_in[18];
  const float* b_ih     = (const float*)d_in[19];
  const float* W_hh     = (const float*)d_in[20];
  const float* b_hh     = (const float*)d_in[21];
  float* out = (float*)d_out;

  size_t off = 0;
  char* base = (char*)d_ws;
  auto carve = [&](size_t bytes) -> char* {
    char* p = base + off;
    off += (bytes + 255) & ~(size_t)255;
    return p;
  };
  int*    order   = (int*)carve(NB * 4);
  int*    lens_s  = (int*)carve(NB * 4);
  float*  enc_s   = (float*)carve((size_t)NB * NE * 4);
  bf16_t* enc_sbf = (bf16_t*)carve((size_t)NB * NE * 2);
  bf16_t* enc_p   = (bf16_t*)carve((size_t)NB * NE * 2);
  bf16_t* hbf     = (bf16_t*)carve((size_t)NB * NE * 2);
  float*  cbuf    = (float*)carve((size_t)NB * NE * 4);
  float*  att1    = (float*)carve((size_t)NB * NE * 4);
  bf16_t* Wdec_p  = (bf16_t*)carve((size_t)NE * NE * 2);
  bf16_t* Wfb_p   = (bf16_t*)carve((size_t)NE * NE * 2);
  bf16_t* Wih_p   = (bf16_t*)carve((size_t)NX * NG * 2);
  bf16_t* Whh_p   = (bf16_t*)carve((size_t)NE * NG * 2);
  bf16_t* Wfc_p   = (bf16_t*)carve((size_t)NE * NV * 2);
  bf16_t* Winh_p  = (bf16_t*)carve((size_t)NE * NE * 2);
  bf16_t* Winc_p  = (bf16_t*)carve((size_t)NE * NE * 2);
  bf16_t* Wenc_p  = (bf16_t*)carve((size_t)NE * NE * 2);

  // 1) sort + gather
  k_order<<<1, NB, 0, stream>>>(caption_lengths, order, lens_s);
  k_gather<<<NB, 256, 0, stream>>>(encoder_out, order, enc_s, enc_sbf);

  // 2) pack weights to WMMA-B layout (bf16)
  auto packNK = [&](const float* W, bf16_t* Bp, int N, int K) {
    size_t total = (size_t)N * K;
    k_pack_nk<<<(unsigned)((total + 255) / 256), 256, 0, stream>>>(W, Bp, N, K);
  };
  packNK(W_dec,    Wdec_p, NE, NE);
  packNK(W_fbeta,  Wfb_p,  NE, NE);
  packNK(W_ih,     Wih_p,  NG, NX);
  packNK(W_hh,     Whh_p,  NG, NE);
  packNK(W_fc,     Wfc_p,  NV, NE);
  packNK(W_init_h, Winh_p, NE, NE);
  packNK(W_init_c, Winc_p, NE, NE);
  packNK(W_enc,    Wenc_p, NE, NE);
  k_pack_kn<<<(unsigned)(((size_t)NB * NE + 255) / 256), 256, 0, stream>>>(
      enc_s, enc_p, NE, NB);

  // 3) initial GEMMs: att1, h0 (bf16), c0 (f32)
  k_gemm_init<<<dim3(8, 4), 256, 0, stream>>>(enc_sbf, Wenc_p, b_enc, att1, nullptr, NE);
  k_gemm_init<<<dim3(8, 4), 256, 0, stream>>>(enc_sbf, Winh_p, b_init_h, nullptr, hbf, NE);
  k_gemm_init<<<dim3(8, 4), 256, 0, stream>>>(enc_sbf, Winc_p, b_init_c, cbuf, nullptr, NE);

  // 4) recurrence
  for (int t = 0; t < NT; ++t) {
    k_step<<<8, 256, K1_SMEM, stream>>>(
        order, lens_s, encoded_captions, emb_table, att1, hbf, cbuf,
        Wdec_p, b_dec, W_full, b_full, enc_p, Wfb_p, b_fbeta,
        Wih_p, b_ih, Whh_p, b_hh, t);
    k_preds<<<dim3(8, 16), 256, 0, stream>>>(hbf, Wfc_p, b_fc, lens_s, out, t);
  }
}
